// ResidualSparseBlock_26783416057880
// MI455X (gfx1250) — matrix operand order repaired and checked by reading, not compiled, not run on hardware
//
#include <hip/hip_runtime.h>
#include <hip/hip_bf16.h>

// ---------------------------------------------------------------------------
// ResidualSparseBlock on MI455X (gfx1250, wave32, WMMA bf16 -> f32 acc)
//
// Roofline: ~57 GFLOP vs ~2.3 GB of random row-gather traffic -> gather bound.
// Keep gather targets bf16 so x (64MB) + h (128MB) ~ fit the 192MB L2, and so
// each gathered row *is* a WMMA A-fragment slice (two b128 loads per lane).
// Streaming flows (idx/mask, f32 x read, f32 out stores) use TH=NT so they
// don't evict the L2-resident gather working set.
// ---------------------------------------------------------------------------

#define CIN  16
#define COUT 32
#define K2   9
#define NKT1 5   // ceil(9/2) K=32 steps for layer 1 (2 offsets * Cin=16 per step)

typedef __attribute__((ext_vector_type(16))) __bf16       v16bf;
typedef __attribute__((ext_vector_type(8)))  float        v8f;
typedef __attribute__((ext_vector_type(4)))  unsigned int u32x4;

union AFrag { v16bf v; u32x4 u[2]; };

static __device__ __forceinline__ u32x4 ld_u32x4(const void* p) {
  return *(const u32x4*)p;
}

// Non-temporal (TH=NT) accessors for single-use streaming data.
static __device__ __forceinline__ int ldnt_i32(const int* p) {
  return __builtin_nontemporal_load(p);
}
static __device__ __forceinline__ unsigned char ldnt_u8(const unsigned char* p) {
  return __builtin_nontemporal_load(p);
}
static __device__ __forceinline__ float ldnt_f32(const float* p) {
  return __builtin_nontemporal_load(p);
}
static __device__ __forceinline__ void stnt_f32(float* p, float v) {
  __builtin_nontemporal_store(v, p);
}

// ---------------------------------------------------------------------------
// Pack weights to bf16 B-fragment-friendly tiles:
//   layout per (ntile, ktile): [16 cols][32 k] bf16 (column-major K runs).
//   Lane L of a wave then loads its B fragment as two 8-bf16 runs (b128 each) at
//   col=(L&15), k=kb..kb+7 and 16+kb..16+kb+7, kb=(L>=16 ? 8 : 0)  -> matches V_WMMA B layout
//   (lanes 0-15 hold K0..7|K16..23 packed 2/VGPR, lanes 16-31 hold K8..15|K24..31).
// ---------------------------------------------------------------------------
__global__ void prep_kernel(const float* __restrict__ W1, const float* __restrict__ b1,
                            const float* __restrict__ W2, const float* __restrict__ b2,
                            const float* __restrict__ Wp, const float* __restrict__ bp,
                            __bf16* __restrict__ bt1, __bf16* __restrict__ bt2,
                            __bf16* __restrict__ btp,
                            float* __restrict__ bias1, float* __restrict__ bias2) {
  const int t = threadIdx.x;
  // layer-1 tiles: [2 ntiles][5 ktiles][16][32]; ktile kt packs offsets (2kt, 2kt+1)
  for (int i = t; i < 2 * NKT1 * 16 * 32; i += blockDim.x) {
    int kk = i & 31;
    int c  = (i >> 5) & 15;
    int kt = (i >> 9) % NKT1;
    int nt = i / (NKT1 * 16 * 32);
    int koff = 2 * kt + (kk >> 4);
    int cin  = kk & 15;
    float v = (koff < K2) ? W1[((size_t)koff * CIN + cin) * COUT + nt * 16 + c] : 0.0f;
    bt1[i] = (__bf16)v;
  }
  // layer-2 tiles: [2][9][16][32]; ktile k covers offset k's full Cout=32 = K
  for (int i = t; i < 2 * K2 * 16 * 32; i += blockDim.x) {
    int kk = i & 31;
    int c  = (i >> 5) & 15;
    int kt = (i >> 9) % K2;
    int nt = i / (K2 * 16 * 32);
    float v = W2[((size_t)kt * COUT + kk) * COUT + nt * 16 + c];
    bt2[i] = (__bf16)v;
  }
  // identity projection tile: [2][16][32], K rows 16..31 zero-padded
  for (int i = t; i < 2 * 16 * 32; i += blockDim.x) {
    int kk = i & 31;
    int c  = (i >> 5) & 15;
    int nt = i / (16 * 32);
    float v = (kk < CIN) ? Wp[(size_t)kk * COUT + nt * 16 + c] : 0.0f;
    btp[i] = (__bf16)v;
  }
  if (t < COUT) {
    bias1[t] = b1[t];
    bias2[t] = b2[t] + bp[t];  // bias of conv2 + bias of identity, fused pre-ReLU
  }
}

// f32 -> bf16 copy of x. Source f32 is read once (NT); the bf16 copy is the
// layer-1/identity gather target and should become L2-resident (RT store).
__global__ void cvt_x_kernel(const float* __restrict__ x, __bf16* __restrict__ xbf, long n) {
  long i = (long)blockIdx.x * blockDim.x + threadIdx.x;
  const long stride = (long)gridDim.x * blockDim.x;
  for (; i < n; i += stride) xbf[i] = (__bf16)ldnt_f32(x + i);
}

// ---------------------------------------------------------------------------
// Layer 1: h = relu(sum_k gather(x)*mask @ W1[k] + b1), one wave per 16 voxels.
// A-fragment (16x32 bf16): lanes 0-15 hold row M=L, K0..7|K16..23; lanes 16-31
// hold row M=L-16, K8..15|K24..31 -> exactly two b128 loads from the two
// gathered bf16 rows (offsets 2kt and 2kt+1) at byte offset (lane>=16 ? 16:0).
// ---------------------------------------------------------------------------
__global__ void layer1_kernel(const __bf16* __restrict__ xbf,
                              const int* __restrict__ nbr_idx,
                              const unsigned char* __restrict__ nbr_mask, // JAX bool_ = 1 byte
                              const __bf16* __restrict__ bt1,
                              const float* __restrict__ bias1,
                              __bf16* __restrict__ hbf, int n) {
  const int lane = threadIdx.x & 31;
  const int wave = threadIdx.x >> 5;
  const int tile = blockIdx.x * (blockDim.x >> 5) + wave;
  const int base = tile * 16;
  if (base >= n) return;              // whole-wave exit: EXEC stays all-ones for WMMA
  const int row  = lane & 15;
  const int g    = base + row;
  const int hi   = lane >> 4;         // 0: K-low half, 1: K-high half
  const int eoff = hi * 8;            // element offset into a 16-elt bf16 row
  const u32x4 z  = {0u, 0u, 0u, 0u};

  v8f c0 = {}; v8f c1 = {};
  #pragma unroll
  for (int kt = 0; kt < NKT1; ++kt) {
    const int k0 = 2 * kt, k1 = 2 * kt + 1;
    AFrag a;
    {
      int  i0 = ldnt_i32(nbr_idx + (long)k0 * n + g);
      bool m0 = ldnt_u8(nbr_mask + (long)k0 * n + g) != 0;
      u32x4 r0 = ld_u32x4(xbf + (long)i0 * CIN + eoff);
      a.u[0] = m0 ? r0 : z;
    }
    if (k1 < K2) {
      int  i1 = ldnt_i32(nbr_idx + (long)k1 * n + g);
      bool m1 = ldnt_u8(nbr_mask + (long)k1 * n + g) != 0;
      u32x4 r1 = ld_u32x4(xbf + (long)i1 * CIN + eoff);
      a.u[1] = m1 ? r1 : z;
    } else {
      a.u[1] = z;                     // zero-padded 5th K-step (offset 9 doesn't exist)
    }

    const __bf16* bpA = bt1 + ((long)kt * 16 + row) * 32 + hi * 8;  // ntile 0
    const __bf16* bpB = bpA + (long)NKT1 * 16 * 32;                  // ntile 1
    AFrag b0, b1f;
    b0.u[0]  = ld_u32x4(bpA);  b0.u[1]  = ld_u32x4(bpA + 16);
    b1f.u[0] = ld_u32x4(bpB);  b1f.u[1] = ld_u32x4(bpB + 16);

    c0 = __builtin_amdgcn_wmma_f32_16x16x32_bf16(false, a.v, false, b0.v,  (short)0, c0, false, false);
    c1 = __builtin_amdgcn_wmma_f32_16x16x32_bf16(false, a.v, false, b1f.v, (short)0, c1, false, false);
  }

  // C/D layout: VGPR r -> (M = r + 8*hi, N = lane&15). Bias + ReLU + bf16 store.
  // hbf is re-read (gathered) by layer 2, so default RT stores keep it in L2.
  const float bb0 = bias1[row];
  const float bb1 = bias1[16 + row];
  #pragma unroll
  for (int r = 0; r < 8; ++r) {
    const long voxel = base + r + hi * 8;
    float v0 = fmaxf(c0[r] + bb0, 0.0f);
    float v1 = fmaxf(c1[r] + bb1, 0.0f);
    hbf[voxel * COUT + row]      = (__bf16)v0;
    hbf[voxel * COUT + 16 + row] = (__bf16)v1;
  }
}

// ---------------------------------------------------------------------------
// Layer 2 + identity: out = relu(sum_k gather(h)*mask @ W2[k] + x@Wp + b2 + bp)
// One offset's 32-channel h row = full K=32 -> two b128 loads per lane:
//   lanes 0-15: bytes [0,16) (K0..7)  and [32,48) (K16..23)
//   lanes16-31: bytes [16,32) (K8..15) and [48,64) (K24..31)
// The identity is a 10th K-step: A = [x_row | 0], B = [Wp ; 0].
// ---------------------------------------------------------------------------
__global__ void layer2_kernel(const __bf16* __restrict__ hbf,
                              const __bf16* __restrict__ xbf,
                              const int* __restrict__ nbr_idx,
                              const unsigned char* __restrict__ nbr_mask,
                              const __bf16* __restrict__ bt2,
                              const __bf16* __restrict__ btp,
                              const float* __restrict__ bias2,
                              float* __restrict__ out, int n) {
  const int lane = threadIdx.x & 31;
  const int wave = threadIdx.x >> 5;
  const int tile = blockIdx.x * (blockDim.x >> 5) + wave;
  const int base = tile * 16;
  if (base >= n) return;
  const int row  = lane & 15;
  const int g    = base + row;
  const int hi   = lane >> 4;
  const int eoff = hi * 8;
  const u32x4 z  = {0u, 0u, 0u, 0u};

  v8f c0 = {}; v8f c1 = {};
  #pragma unroll
  for (int k = 0; k < K2; ++k) {
    int  i0 = ldnt_i32(nbr_idx + (long)k * n + g);
    bool m  = ldnt_u8(nbr_mask + (long)k * n + g) != 0;
    const __bf16* hr = hbf + (long)i0 * COUT;
    AFrag a;
    u32x4 rlo = ld_u32x4(hr + eoff);        // K 0..7  / 8..15
    u32x4 rhi = ld_u32x4(hr + 16 + eoff);   // K16..23 / 24..31
    a.u[0] = m ? rlo : z;
    a.u[1] = m ? rhi : z;

    const __bf16* bpA = bt2 + ((long)k * 16 + row) * 32 + hi * 8;  // ntile 0
    const __bf16* bpB = bpA + (long)K2 * 16 * 32;                   // ntile 1
    AFrag b0, b1f;
    b0.u[0]  = ld_u32x4(bpA);  b0.u[1]  = ld_u32x4(bpA + 16);
    b1f.u[0] = ld_u32x4(bpB);  b1f.u[1] = ld_u32x4(bpB + 16);

    c0 = __builtin_amdgcn_wmma_f32_16x16x32_bf16(false, a.v, false, b0.v,  (short)0, c0, false, false);
    c1 = __builtin_amdgcn_wmma_f32_16x16x32_bf16(false, a.v, false, b1f.v, (short)0, c1, false, false);
  }

  // identity K-step: A rows = x[g] (16 ch, K high half zero), B = padded Wp tile
  {
    AFrag a;
    a.u[0] = ld_u32x4(xbf + (long)g * CIN + eoff);  // K0..7 / K8..15 from x
    a.u[1] = z;                                      // K16..31 zero
    const __bf16* bpA = btp + (long)row * 32 + hi * 8;
    const __bf16* bpB = bpA + 16 * 32;
    AFrag b0, b1f;
    b0.u[0]  = ld_u32x4(bpA);  b0.u[1]  = ld_u32x4(bpA + 16);
    b1f.u[0] = ld_u32x4(bpB);  b1f.u[1] = ld_u32x4(bpB + 16);
    c0 = __builtin_amdgcn_wmma_f32_16x16x32_bf16(false, a.v, false, b0.v,  (short)0, c0, false, false);
    c1 = __builtin_amdgcn_wmma_f32_16x16x32_bf16(false, a.v, false, b1f.v, (short)0, c1, false, false);
  }

  // Output is written once and never re-read: NT stores keep xbf/hbf in L2.
  const float bb0 = bias2[row];
  const float bb1 = bias2[16 + row];
  #pragma unroll
  for (int r = 0; r < 8; ++r) {
    const long voxel = base + r + hi * 8;
    stnt_f32(out + voxel * COUT + row,      fmaxf(c0[r] + bb0, 0.0f));
    stnt_f32(out + voxel * COUT + 16 + row, fmaxf(c1[r] + bb1, 0.0f));
  }
}

// ---------------------------------------------------------------------------
extern "C" void kernel_launch(void* const* d_in, const int* in_sizes, int n_in,
                              void* d_out, int out_size, void* d_ws, size_t ws_size,
                              hipStream_t stream) {
  const float* x  = (const float*)d_in[0];
  const float* W1 = (const float*)d_in[1];
  const float* b1 = (const float*)d_in[2];
  const float* W2 = (const float*)d_in[3];
  const float* b2 = (const float*)d_in[4];
  const float* Wp = (const float*)d_in[5];
  const float* bp = (const float*)d_in[6];
  const int*   nbr_idx  = (const int*)d_in[7];
  const unsigned char* nbr_mask = (const unsigned char*)d_in[8];  // jnp.bool_ = 1 byte

  const int n = in_sizes[0] / CIN;   // N = 2,000,000 (divisible by 16)

  // workspace carve-out (256B aligned slices)
  char* w = (char*)d_ws;
  size_t off = 0;
  auto take = [&](size_t bytes) -> char* {
    char* p = w + off;
    off = (off + bytes + 255) & ~(size_t)255;
    return p;
  };
  __bf16* xbf  = (__bf16*)take((size_t)n * CIN * sizeof(__bf16));   // 64 MB
  __bf16* hbf  = (__bf16*)take((size_t)n * COUT * sizeof(__bf16));  // 128 MB
  __bf16* bt1  = (__bf16*)take((size_t)2 * NKT1 * 16 * 32 * sizeof(__bf16));
  __bf16* bt2  = (__bf16*)take((size_t)2 * K2 * 16 * 32 * sizeof(__bf16));
  __bf16* btp  = (__bf16*)take((size_t)2 * 16 * 32 * sizeof(__bf16));
  float*  bias1 = (float*)take(COUT * sizeof(float));
  float*  bias2 = (float*)take(COUT * sizeof(float));

  prep_kernel<<<1, 256, 0, stream>>>(W1, b1, W2, b2, Wp, bp, bt1, bt2, btp, bias1, bias2);
  cvt_x_kernel<<<2048, 256, 0, stream>>>(x, xbf, (long)n * CIN);

  const int tiles  = n / 16;                 // one wave32 per 16-voxel tile
  const int blocks = (tiles + 7) / 8;        // 256 threads = 8 waves per block
  layer1_kernel<<<blocks, 256, 0, stream>>>(xbf, nbr_idx, nbr_mask, bt1, bias1, hbf, n);
  layer2_kernel<<<blocks, 256, 0, stream>>>(hbf, xbf, nbr_idx, nbr_mask, bt2, btp, bias2,
                                            (float*)d_out, n);
}
